// Token_performer_49976239456366
// MI455X (gfx1250) — matrix operand order repaired and checked by reading, not compiled, not run on hardware
//
#include <hip/hip_runtime.h>
#include <hip/hip_bf16.h>
#include <math.h>

typedef _Float16 half_t;
typedef __attribute__((ext_vector_type(16))) _Float16 v16h;
typedef __attribute__((ext_vector_type(8)))  _Float16 v8h;
typedef __attribute__((ext_vector_type(8)))  float    v8f;

#define DIMF   147
#define DIMP   160      // DIM padded to multiple of 32
#define EMB    64
#define N3     192      // 3*EMB
#define MF     32       // random-feature dim
#define BATCH  32
#define TSEQ   3136
#define BT     (BATCH*TSEQ)   // 100352

// ---------------- WMMA fragment helpers (contiguous k = 16*(lane/16) + s in A and B) ----------------

// A-fragment 16x32 f16 from row-major [*, lda] at (row0, k0).
__device__ __forceinline__ v16h load_a_frag(const half_t* A, int lda, int row0, int k0, int lane) {
  int m  = lane & 15;
  int ks = k0 + ((lane >> 4) << 4);
  const half_t* p = A + (size_t)(row0 + m) * lda + ks;
  v8h lo = *(const v8h*)p;
  v8h hi = *(const v8h*)(p + 8);
  v16h r;
#pragma unroll
  for (int i = 0; i < 8; ++i) { r[i] = lo[i]; r[i + 8] = hi[i]; }
  return r;
}

// B-fragment 32x16 f16 from B stored [N][K] row-major (ldb = K), tile (n0, k0).
__device__ __forceinline__ v16h load_b_frag(const half_t* B, int ldb, int n0, int k0, int lane) {
  int n  = n0 + (lane & 15);
  int ks = k0 + ((lane >> 4) << 4);
  return *(const v16h*)(B + (size_t)n * ldb + ks);
}

__device__ __forceinline__ v8f wmma_f16(v16h a, v16h b, v8f c) {
  return __builtin_amdgcn_wmma_f32_16x16x32_f16(false, a, false, b, (short)0, c, false, false);
}

__device__ __forceinline__ float wave_allreduce(float v) {
#pragma unroll
  for (int m = 1; m < 32; m <<= 1) v += __shfl_xor(v, m, 32);
  return v;
}

// Async global->LDS copy of one 16B chunk (CDNA5 async path, tracked by ASYNCcnt).
__device__ __forceinline__ void async_copy_b128(unsigned int lds_addr, const void* gptr) {
  unsigned long long ga = (unsigned long long)(uintptr_t)gptr;
  asm volatile("global_load_async_to_lds_b128 %0, %1, off" :: "v"(lds_addr), "v"(ga) : "memory");
}
__device__ __forceinline__ void async_wait0() {
  asm volatile("s_wait_asynccnt 0x0" ::: "memory");
}

// ---------------- Kernel 1: weight prep (f16 [N][K] layouts) + zero kpsum ----------------
__global__ void prep_kernel(const float* w_kqv, const float* W_rf, const float* w_proj,
                            const float* w_m1, const float* w_m2,
                            half_t* Bk, half_t* Brf, half_t* Bp, half_t* Bm1, half_t* Bm2,
                            float* kpsum) {
  int idx = blockIdx.x * 256 + threadIdx.x;
  if (idx < N3 * DIMP) {
    int n = idx / DIMP, k = idx % DIMP;
    Bk[idx] = (half_t)(k < DIMF ? w_kqv[n * DIMF + k] : 0.f);  // zero-pad K
  }
  if (idx < MF * EMB)  Brf[idx] = (half_t)W_rf[idx];           // W_rf is [m][e] == [N][K]
  if (idx < EMB * EMB) {
    Bp[idx]  = (half_t)w_proj[idx];                            // [n][e] natural
    Bm1[idx] = (half_t)w_m1[idx];
    Bm2[idx] = (half_t)w_m2[idx];
  }
  if (idx < BATCH * MF) kpsum[idx] = 0.f;
}

// ---------------- Kernel 2: LayerNorm over DIM=147, write f16 padded to 160 ----------------
__global__ void ln1_kernel(const float* __restrict__ x, const float* __restrict__ g,
                           const float* __restrict__ b, half_t* __restrict__ xn) {
  int lane = threadIdx.x & 31;
  int row  = blockIdx.x * 8 + (threadIdx.x >> 5);
  float v[5], s = 0.f, ss = 0.f;
#pragma unroll
  for (int i = 0; i < 5; ++i) {
    int idx = lane + 32 * i;
    float xv = (idx < DIMF) ? x[(size_t)row * DIMF + idx] : 0.f;
    v[i] = xv; s += xv; ss += xv * xv;
  }
  s = wave_allreduce(s); ss = wave_allreduce(ss);
  float mu = s * (1.f / DIMF);
  float rs = rsqrtf(ss * (1.f / DIMF) - mu * mu + 1e-5f);
#pragma unroll
  for (int i = 0; i < 5; ++i) {
    int idx = lane + 32 * i;
    half_t o = (half_t)0.f;
    if (idx < DIMF) o = (half_t)((v[i] - mu) * rs * g[idx] + b[idx]);
    xn[(size_t)row * DIMP + idx] = o;
  }
}

// ---------------- Kernel 3: kqv = xn @ Wkqv^T + b  (WMMA, K=160) ----------------
// B-panel (64 x 160 f16 = 20KB) staged into LDS via async global->LDS loads.
__global__ void kqv_kernel(const half_t* __restrict__ xn, const half_t* __restrict__ Bk,
                           const float* __restrict__ b_kqv,
                           half_t* __restrict__ kq, float* __restrict__ vf,
                           half_t* __restrict__ vT) {
  __shared__ half_t Bs[64][DIMP];                     // 20480 B
  int lane = threadIdx.x & 31, w = threadIdx.x >> 5;

  // Async-stage this block's B panel into LDS (1280 x 16B chunks, 5 per thread).
  {
    const char* Bg = (const char*)(Bk + (size_t)(blockIdx.y * 64) * DIMP);
    unsigned int lbase = (unsigned int)(uintptr_t)(&Bs[0][0]);
#pragma unroll
    for (int i = 0; i < 5; ++i) {
      int c = threadIdx.x + i * 256;                  // chunk id, 1280 total
      async_copy_b128(lbase + c * 16, Bg + (size_t)c * 16);
    }
    async_wait0();
    __syncthreads();
  }

  int row0 = blockIdx.x * 32 + (w >> 2) * 16;
  int n0l  = (w & 3) * 16;                            // n-tile inside LDS panel
  v8f acc = {};
#pragma unroll
  for (int kk = 0; kk < DIMP; kk += 32)
    acc = wmma_f16(load_a_frag(xn, DIMP, row0, kk, lane),
                   load_b_frag(&Bs[0][0], DIMP, n0l, kk, lane), acc);

  int n = blockIdx.y * 64 + n0l + (lane & 15);
  float bias = b_kqv[n];
  int base = (lane >> 4) << 3;
  int bb = row0 / TSEQ;                               // 32-bit uniform div (scalar magic-mul)
  int t0 = row0 - bb * TSEQ + base;
#pragma unroll
  for (int r = 0; r < 8; ++r) {
    int row = row0 + base + r;
    float val = acc[r] + bias;
    if (n < 2 * EMB) {
      kq[(size_t)row * 128 + n] = (half_t)val;
    } else {
      int e = n - 128;
      vf[(size_t)row * EMB + e] = val;
      vT[((size_t)bb * EMB + e) * TSEQ + (t0 + r)] = (half_t)val;
    }
  }
}

// ---------------- Kernel 4: prm_exp (WMMA K=64) + optional kp batch-sum ----------------
// mode 0: k-path -> kpT [B][MF][T] f16 + atomic kpsum ; mode 1: q-path -> qp [BT][MF] f16
__global__ void prm_kernel(const half_t* __restrict__ kq, int koff,
                           const half_t* __restrict__ Brf, int mode,
                           half_t* __restrict__ kpT, half_t* __restrict__ qp,
                           float* __restrict__ kpsum) {
  int lane = threadIdx.x & 31, w = threadIdx.x >> 5;
  int row0 = blockIdx.x * 64 + (w >> 1) * 16;
  int n0   = (w & 1) * 16;
  // xd = 0.5*||row||^2 : lane handles row (lane&15), half (lane>>4) of 64 cols
  int r15 = lane & 15, sel = lane >> 4;
  const half_t* src = kq + (size_t)(row0 + r15) * 128 + koff + sel * 32;
  float ssq = 0.f;
#pragma unroll
  for (int i = 0; i < 32; ++i) { float t = (float)src[i]; ssq += t * t; }
  ssq += __shfl_xor(ssq, 16, 32);
  ssq *= 0.5f;
  int base = sel << 3;
  float xdv[8];
#pragma unroll
  for (int r = 0; r < 8; ++r) xdv[r] = __shfl(ssq, base + r, 32);  // row held by lane==row

  v8f acc = {};
#pragma unroll
  for (int kk = 0; kk < EMB; kk += 32)
    acc = wmma_f16(load_a_frag(kq + koff, 128, row0, kk, lane),
                   load_b_frag(Brf, EMB, n0, kk, lane), acc);

  int n = n0 + (lane & 15);
  float lsum = 0.f;
  int bb = row0 / TSEQ;                 // 16-row tile never crosses a batch (3136 % 16 == 0)
  int t0 = row0 - bb * TSEQ + base;
#pragma unroll
  for (int r = 0; r < 8; ++r) {
    float val = expf(acc[r] - xdv[r]) * 0.17677669529663687f;  // 1/sqrt(32)
    if (mode == 0) {
      kpT[((size_t)bb * MF + n) * TSEQ + (t0 + r)] = (half_t)val;
      lsum += val;
    } else {
      qp[(size_t)(row0 + base + r) * MF + n] = (half_t)val;
    }
  }
  if (mode == 0) {
    lsum += __shfl_xor(lsum, 16, 32);
    if (lane < 16) atomicAdd(&kpsum[bb * MF + n], lsum);
  }
}

// ---------------- Kernel 5: kptv[b,e,m] = sum_t v[b,t,e] * kp[b,t,m]  (WMMA K=T) ----------
__global__ void kptv_kernel(const half_t* __restrict__ vT, const half_t* __restrict__ kpT,
                            half_t* __restrict__ kptv) {
  int lane = threadIdx.x & 31, w = threadIdx.x >> 5;
  int bb = blockIdx.x;
  int e0 = (w >> 1) << 4, n0 = (w & 1) << 4;
  const half_t* A = vT  + (size_t)bb * EMB * TSEQ;  // [e][t]
  const half_t* B = kpT + (size_t)bb * MF  * TSEQ;  // [m][t]
  const half_t* arow = A + (size_t)(e0 + (lane & 15)) * TSEQ;
  const half_t* brow = B + (size_t)(n0 + (lane & 15)) * TSEQ;
  v8f acc = {};
  for (int kk = 0; kk < TSEQ; kk += 32) {
    if (kk + 512 < TSEQ) {                           // stream-ahead: global_prefetch_b8
      __builtin_prefetch(arow + kk + 512, 0, 1);
      __builtin_prefetch(brow + kk + 512, 0, 1);
    }
    acc = wmma_f16(load_a_frag(A, TSEQ, e0, kk, lane),
                   load_b_frag(B, TSEQ, n0, kk, lane), acc);
  }
  int mm = n0 + (lane & 15);
  int base = (lane >> 4) << 3;
#pragma unroll
  for (int r = 0; r < 8; ++r) {
    int e = e0 + base + r;
    kptv[((size_t)bb * EMB + e) * MF + mm] = (half_t)acc[r];   // [b][e][m] == B-layout [N][K]
  }
}

// ---------------- Kernel 6: y = v + proj(qp @ kptv^T / (D+eps)) + b_proj ----------------
__global__ void attn_kernel(const half_t* __restrict__ qp, const half_t* __restrict__ kptv,
                            const float* __restrict__ kpsum, const half_t* __restrict__ Bp,
                            const float* __restrict__ b_proj, const float* __restrict__ vf,
                            float* __restrict__ y) {
  __shared__ half_t yat[8][16][72];
  int lane = threadIdx.x & 31, w = threadIdx.x >> 5;
  int row0 = blockIdx.x * 128 + w * 16;
  int bb = row0 / TSEQ;
  int r15 = lane & 15, sel = lane >> 4, base = sel << 3;
  // D[t] = qp[t,:] . kpsum[b,:]
  const half_t* qrow = qp + (size_t)(row0 + r15) * MF + sel * 16;
  const float*  ksum = kpsum + bb * MF + sel * 16;
  float d = 0.f;
#pragma unroll
  for (int i = 0; i < 16; ++i) d += (float)qrow[i] * ksum[i];
  d += __shfl_xor(d, 16, 32);
  float invD[8];
#pragma unroll
  for (int r = 0; r < 8; ++r) invD[r] = 1.f / (__shfl(d, base + r, 32) + 1e-8f);

  // y_attn tile (16 tokens x 64 e), K = MF = 32 (single WMMA step)
  v16h afrag = load_a_frag(qp, MF, row0, 0, lane);
  const half_t* Bkv = kptv + (size_t)bb * EMB * MF;
#pragma unroll
  for (int e0 = 0; e0 < EMB; e0 += 16) {
    v8f acc = {};
    acc = wmma_f16(afrag, load_b_frag(Bkv, MF, e0, 0, lane), acc);
    int n = e0 + (lane & 15);
#pragma unroll
    for (int r = 0; r < 8; ++r) yat[w][base + r][n] = (half_t)(acc[r] * invD[r]);
  }
  __syncthreads();
  // proj: [16x64] @ w_proj^T, K = 64
  const half_t* AL = &yat[w][0][0];
#pragma unroll
  for (int e0 = 0; e0 < EMB; e0 += 16) {
    v8f acc = {};
#pragma unroll
    for (int kk = 0; kk < EMB; kk += 32)
      acc = wmma_f16(load_a_frag(AL, 72, 0, kk, lane),
                     load_b_frag(Bp, EMB, e0, kk, lane), acc);
    int n = e0 + (lane & 15);
    float bp = b_proj[n];
#pragma unroll
    for (int r = 0; r < 8; ++r) {
      size_t row = row0 + base + r;
      y[row * EMB + n] = acc[r] + bp + vf[row * EMB + n];  // residual on v
    }
  }
}

// ---------------- Kernel 7: LayerNorm over EMB=64 -> f16 ----------------
__global__ void ln2_kernel(const float* __restrict__ y, const float* __restrict__ g,
                           const float* __restrict__ b, half_t* __restrict__ yn) {
  int lane = threadIdx.x & 31;
  int row  = blockIdx.x * 8 + (threadIdx.x >> 5);
  float v0 = y[(size_t)row * EMB + lane];
  float v1 = y[(size_t)row * EMB + lane + 32];
  float s  = wave_allreduce(v0 + v1);
  float ss = wave_allreduce(v0 * v0 + v1 * v1);
  float mu = s * (1.f / EMB);
  float rs = rsqrtf(ss * (1.f / EMB) - mu * mu + 1e-5f);
  yn[(size_t)row * EMB + lane]      = (half_t)((v0 - mu) * rs * g[lane]      + b[lane]);
  yn[(size_t)row * EMB + lane + 32] = (half_t)((v1 - mu) * rs * g[lane + 32] + b[lane + 32]);
}

// ---------------- Kernels 8/9: MLP GEMMs (K=64, N=64). mode 0: +gelu -> f16. mode 1: +resid -> f32
__global__ void mlp_kernel(const half_t* __restrict__ A, const half_t* __restrict__ Bw,
                           const float* __restrict__ bias, const float* __restrict__ resid,
                           half_t* __restrict__ outH, float* __restrict__ outF, int mode) {
  int lane = threadIdx.x & 31, w = threadIdx.x >> 5;
  int row0 = blockIdx.x * 32 + (w >> 2) * 16;
  int n0   = (w & 3) * 16;
  v8f acc = {};
#pragma unroll
  for (int kk = 0; kk < EMB; kk += 32)
    acc = wmma_f16(load_a_frag(A, EMB, row0, kk, lane),
                   load_b_frag(Bw, EMB, n0, kk, lane), acc);
  int n = n0 + (lane & 15);
  float bv = bias[n];
  int base = (lane >> 4) << 3;
#pragma unroll
  for (int r = 0; r < 8; ++r) {
    size_t row = row0 + base + r;
    float val = acc[r] + bv;
    if (mode == 0) {
      float gl = 0.5f * val * (1.f + erff(val * 0.70710678118654752f));  // exact gelu
      outH[row * EMB + n] = (half_t)gl;
    } else {
      outF[row * EMB + n] = val + resid[row * EMB + n];
    }
  }
}

// ---------------- Host launch ----------------
extern "C" void kernel_launch(void* const* d_in, const int* in_sizes, int n_in,
                              void* d_out, int out_size, void* d_ws, size_t ws_size,
                              hipStream_t stream) {
  const float* x      = (const float*)d_in[0];
  const float* w_kqv  = (const float*)d_in[1];
  const float* b_kqv  = (const float*)d_in[2];
  const float* w_proj = (const float*)d_in[3];
  const float* b_proj = (const float*)d_in[4];
  const float* g1     = (const float*)d_in[5];
  const float* beta1  = (const float*)d_in[6];
  const float* g2     = (const float*)d_in[7];
  const float* beta2  = (const float*)d_in[8];
  const float* w_m1   = (const float*)d_in[9];
  const float* b_m1   = (const float*)d_in[10];
  const float* w_m2   = (const float*)d_in[11];
  const float* b_m2   = (const float*)d_in[12];
  const float* W_rf   = (const float*)d_in[13];
  float* out = (float*)d_out;

  char* ws = (char*)d_ws;
  size_t off = 0;
  auto alloc = [&](size_t bytes) { size_t o = off; off += (bytes + 255) & ~(size_t)255; return o; };
  size_t off_xn   = alloc((size_t)BT * DIMP * 2);           // xn f16 (reused later for yn/h)
  size_t off_kq   = alloc((size_t)BT * 128 * 2);            // k,q f16
  size_t off_vf   = alloc((size_t)BT * EMB * 4);            // v f32
  size_t off_vT   = alloc((size_t)BATCH * EMB * TSEQ * 2);  // v^T f16
  size_t off_qp   = alloc((size_t)BT * MF * 2);             // qp f16
  size_t off_kpT  = alloc((size_t)BATCH * MF * TSEQ * 2);   // kp^T f16
  size_t off_ksum = alloc((size_t)BATCH * MF * 4);          // kp batch sums f32
  size_t off_kptv = alloc((size_t)BATCH * EMB * MF * 2);    // kptv f16
  size_t off_y    = alloc((size_t)BT * EMB * 4);            // y f32
  size_t off_Bk   = alloc((size_t)N3 * DIMP * 2);
  size_t off_Brf  = alloc((size_t)MF * EMB * 2);
  size_t off_Bp   = alloc((size_t)EMB * EMB * 2);
  size_t off_Bm1  = alloc((size_t)EMB * EMB * 2);
  size_t off_Bm2  = alloc((size_t)EMB * EMB * 2);
  (void)ws_size; (void)in_sizes; (void)n_in; (void)out_size;

  half_t* xn   = (half_t*)(ws + off_xn);
  half_t* kq   = (half_t*)(ws + off_kq);
  float*  vf   = (float*) (ws + off_vf);
  half_t* vT   = (half_t*)(ws + off_vT);
  half_t* qp   = (half_t*)(ws + off_qp);
  half_t* kpT  = (half_t*)(ws + off_kpT);
  float*  ksum = (float*) (ws + off_ksum);
  half_t* kptv = (half_t*)(ws + off_kptv);
  float*  y    = (float*) (ws + off_y);
  half_t* Bk   = (half_t*)(ws + off_Bk);
  half_t* Brf  = (half_t*)(ws + off_Brf);
  half_t* Bp   = (half_t*)(ws + off_Bp);
  half_t* Bm1  = (half_t*)(ws + off_Bm1);
  half_t* Bm2  = (half_t*)(ws + off_Bm2);
  half_t* yn   = (half_t*)(ws + off_xn);                      // reuse xn region
  half_t* h    = (half_t*)(ws + off_xn + (size_t)BT * EMB * 2);

  prep_kernel<<<(N3 * DIMP + 255) / 256, 256, 0, stream>>>(
      w_kqv, W_rf, w_proj, w_m1, w_m2, Bk, Brf, Bp, Bm1, Bm2, ksum);
  ln1_kernel<<<BT / 8, 256, 0, stream>>>(x, g1, beta1, xn);
  kqv_kernel<<<dim3(BT / 32, N3 / 64), 256, 0, stream>>>(xn, Bk, b_kqv, kq, vf, vT);
  prm_kernel<<<BT / 64, 256, 0, stream>>>(kq, 0,   Brf, 0, kpT, qp, ksum);  // k-path
  prm_kernel<<<BT / 64, 256, 0, stream>>>(kq, EMB, Brf, 1, kpT, qp, ksum);  // q-path
  kptv_kernel<<<BATCH, 256, 0, stream>>>(vT, kpT, kptv);
  attn_kernel<<<BT / 128, 256, 0, stream>>>(qp, kptv, ksum, Bp, b_proj, vf, y);
  ln2_kernel<<<BT / 8, 256, 0, stream>>>(y, g2, beta2, yn);
  mlp_kernel<<<BT / 32, 256, 0, stream>>>(yn, Bm1, b_m1, nullptr, h, nullptr, 0);
  mlp_kernel<<<BT / 32, 256, 0, stream>>>(h,  Bm2, b_m2, y, nullptr, out, 1);
}
